// DeformableTransformerEncoderLayer_7541962572418
// MI455X (gfx1250) — matrix operand order repaired and checked by reading, not compiled, not run on hardware
//
#include <hip/hip_runtime.h>
#include <hip/hip_bf16.h>

typedef _Float16 v16h __attribute__((ext_vector_type(16)));
typedef _Float16 v8h  __attribute__((ext_vector_type(8)));
typedef float    v8f  __attribute__((ext_vector_type(8)));

union AFrag { v16h v; v8h h[2]; };
union CFrag { v8f  v; float f[8]; };

#define BM 128
#define BN 64
#define TK 32
#define LSTRIDE 40      // TK + 8 pad (halves): 80B row stride, conflict-free b128
#define USE_ASYNC_LDS 1 // gfx1250 GLOBAL_LOAD_ASYNC_TO_LDS path (ASYNCcnt)

// smem layout (halves), single symbol so index math is exact:
//   A bufs: [2][BM][LSTRIDE]  at 0
//   B bufs: [2][BN][LSTRIDE]  at 2*BM*LSTRIDE
#define SA_OFF(buf, r, c) (((buf) * BM + (r)) * LSTRIDE + (c))
#define SB_OFF(buf, r, c) (2 * BM * LSTRIDE + ((buf) * BN + (r)) * LSTRIDE + (c))
#define SMEM_HALVES (2 * BM * LSTRIDE + 2 * BN * LSTRIDE)

__device__ __forceinline__ void async_ld16(const _Float16* g, const _Float16* l)
{
    // CDNA5 async copy global->LDS, 16B per lane, tracked by ASYNCcnt.
    // LDS dest VGPR = wave-relative LDS byte offset = low 32 bits of the
    // generic shared-aperture address (ISA 10.2).
    asm volatile("global_load_async_to_lds_b128 %0, %1, off"
                 :: "v"((unsigned)(uintptr_t)l), "v"(g)
                 : "memory");
}
__device__ __forceinline__ void async_wait0()
{
    asm volatile("s_wait_asynccnt 0x0" ::: "memory");
}

// ---------------------------------------------------------------------------
// WMMA GEMM: C[M,N] = A[M,K] (f16 row-major) x Bw[N,K]^T (f16) + bias
// 256 threads = 8 waves, block tile 128x64, wave tile 32x32 (4 WMMA / K-step),
// double-buffered LDS fed by async-to-LDS copies, one barrier per K-step.
// ---------------------------------------------------------------------------
template<bool RELU, bool WF32, bool WF16>
__global__ __launch_bounds__(256)
void gemm_wmma_kernel(const _Float16* __restrict__ A,
                      const _Float16* __restrict__ Bw,
                      const float*    __restrict__ bias,
                      float*          __restrict__ C,
                      _Float16*       __restrict__ Ch,
                      int M, int N, int K)
{
    __shared__ _Float16 smem[SMEM_HALVES];

    const int tid  = threadIdx.x;
    const int lane = tid & 31;
    const int wave = tid >> 5;
    const int wm   = wave & 3;   // 32-row slab within block tile
    const int wn   = wave >> 2;  // 32-col slab within block tile

    const int m0 = blockIdx.y * BM;
    const int n0 = blockIdx.x * BN;

    // staging: 16B chunks; A tile needs 512 chunks (2/thread), B tile 256 (1/thread)
    const int sr  = tid >> 2;         // 0..63
    const int sc8 = (tid & 3) * 8;    // 0,8,16,24

    const int ar0 = min(m0 + sr, M - 1);
    const int ar1 = min(m0 + sr + 64, M - 1);
    const int brc = min(n0 + sr, N - 1);   // clamp: cols >= N computed, never stored
    const size_t aBase0 = (size_t)ar0 * K + sc8;
    const size_t aBase1 = (size_t)ar1 * K + sc8;
    const size_t bBase  = (size_t)brc * K + sc8;

    CFrag acc[2][2];
    #pragma unroll
    for (int mi = 0; mi < 2; ++mi)
        #pragma unroll
        for (int ni = 0; ni < 2; ++ni)
            #pragma unroll
            for (int i = 0; i < 8; ++i) acc[mi][ni].f[i] = 0.f;

    const int nk = K / TK;

#if USE_ASYNC_LDS
    async_ld16(A  + aBase0, &smem[SA_OFF(0, sr,      sc8)]);
    async_ld16(A  + aBase1, &smem[SA_OFF(0, sr + 64, sc8)]);
    async_ld16(Bw + bBase,  &smem[SB_OFF(0, sr,      sc8)]);
    async_wait0();
    __syncthreads();
#else
    {
        v8h ra0 = *(const v8h*)(A + aBase0);
        v8h ra1 = *(const v8h*)(A + aBase1);
        v8h rb  = *(const v8h*)(Bw + bBase);
        *(v8h*)&smem[SA_OFF(0, sr,      sc8)] = ra0;
        *(v8h*)&smem[SA_OFF(0, sr + 64, sc8)] = ra1;
        *(v8h*)&smem[SB_OFF(0, sr,      sc8)] = rb;
        __syncthreads();
    }
#endif

    for (int kk = 0; kk < nk; ++kk) {
        const int cur = kk & 1;
        const int nxt = cur ^ 1;
        const bool more = (kk + 1) < nk;
        const size_t ko = (size_t)(kk + 1) * TK;

#if USE_ASYNC_LDS
        if (more) {
            async_ld16(A  + aBase0 + ko, &smem[SA_OFF(nxt, sr,      sc8)]);
            async_ld16(A  + aBase1 + ko, &smem[SA_OFF(nxt, sr + 64, sc8)]);
            async_ld16(Bw + bBase  + ko, &smem[SB_OFF(nxt, sr,      sc8)]);
        }
#else
        v8h ra0, ra1, rb;
        if (more) {
            ra0 = *(const v8h*)(A + aBase0 + ko);
            ra1 = *(const v8h*)(A + aBase1 + ko);
            rb  = *(const v8h*)(Bw + bBase + ko);
        }
#endif

        // A fragments: 16-bit A 16x32 layout (lanes 0-15: K 0-7/16-23;
        // lanes 16-31: K 8-15/24-31)
        AFrag a0, a1, b0, b1;
        const int arow = (wm << 5) + (lane & 15);
        const int akb  = (lane < 16) ? 0 : 8;
        a0.h[0] = *(const v8h*)&smem[SA_OFF(cur, arow,      akb)];
        a0.h[1] = *(const v8h*)&smem[SA_OFF(cur, arow,      akb + 16)];
        a1.h[0] = *(const v8h*)&smem[SA_OFF(cur, arow + 16, akb)];
        a1.h[1] = *(const v8h*)&smem[SA_OFF(cur, arow + 16, akb + 16)];

        // B fragments: lane = column n; lanes 0-15 carry K 0-7/16-23, 16-31 carry K 8-15/24-31
        const int brow = (wn << 5) + (lane & 15);
        const int bkb  = (lane < 16) ? 0 : 8;
        b0.h[0] = *(const v8h*)&smem[SB_OFF(cur, brow,      bkb)];
        b0.h[1] = *(const v8h*)&smem[SB_OFF(cur, brow,      bkb + 16)];
        b1.h[0] = *(const v8h*)&smem[SB_OFF(cur, brow + 16, bkb)];
        b1.h[1] = *(const v8h*)&smem[SB_OFF(cur, brow + 16, bkb + 16)];

        acc[0][0].v = __builtin_amdgcn_wmma_f32_16x16x32_f16(false, a0.v, false, b0.v,
                                                             (short)0, acc[0][0].v, false, false);
        acc[0][1].v = __builtin_amdgcn_wmma_f32_16x16x32_f16(false, a0.v, false, b1.v,
                                                             (short)0, acc[0][1].v, false, false);
        acc[1][0].v = __builtin_amdgcn_wmma_f32_16x16x32_f16(false, a1.v, false, b0.v,
                                                             (short)0, acc[1][0].v, false, false);
        acc[1][1].v = __builtin_amdgcn_wmma_f32_16x16x32_f16(false, a1.v, false, b1.v,
                                                             (short)0, acc[1][1].v, false, false);

#if USE_ASYNC_LDS
        async_wait0();
        __syncthreads();
#else
        __syncthreads();
        if (more) {
            *(v8h*)&smem[SA_OFF(nxt, sr,      sc8)] = ra0;
            *(v8h*)&smem[SA_OFF(nxt, sr + 64, sc8)] = ra1;
            *(v8h*)&smem[SB_OFF(nxt, sr,      sc8)] = rb;
        }
        __syncthreads();
#endif
    }

    // Epilogue: D layout — VGPR i: lanes 0-15 -> M=i, lanes 16-31 -> M=8+i; N=lane%16
    #pragma unroll
    for (int mi = 0; mi < 2; ++mi) {
        #pragma unroll
        for (int ni = 0; ni < 2; ++ni) {
            const int mbase = m0 + (wm << 5) + (mi << 4) + ((lane >> 4) << 3);
            const int nn    = n0 + (wn << 5) + (ni << 4) + (lane & 15);
            if (nn < N) {
                const float bn = bias[nn];
                #pragma unroll
                for (int i = 0; i < 8; ++i) {
                    const int m = mbase + i;
                    if (m < M) {
                        float v = acc[mi][ni].f[i] + bn;
                        if (RELU) v = fmaxf(v, 0.f);
                        if constexpr (WF32) C [(size_t)m * N + nn] = v;
                        if constexpr (WF16) Ch[(size_t)m * N + nn] = (_Float16)v;
                    }
                }
            }
        }
    }
}

// ---------------------------------------------------------------------------
// Deformable attention sampling: one wave per (token, head), lane = channel.
// ---------------------------------------------------------------------------
__global__ __launch_bounds__(256)
void msda_sample_kernel(const float* __restrict__ value,  // [B*L, 256]
                        const float* __restrict__ off,    // [B*L, 64]
                        const float* __restrict__ attn,   // [B*L, 32] logits
                        _Float16*    __restrict__ outh,   // [B*L, 256]
                        int Bb, int Hh, int Ww)
{
    const int gid  = blockIdx.x * blockDim.x + threadIdx.x;
    const int lane = gid & 31;
    const int wid  = gid >> 5;
    const int t    = wid >> 3;     // token
    const int hh   = wid & 7;      // head
    const int L = Hh * Ww;
    if (t >= Bb * L) return;
    const int b  = t / L;
    const int l  = t - b * L;
    const int iy = l / Ww;
    const int ix = l - iy * Ww;

    const float* aw = attn + (size_t)t * 32 + hh * 4;
    const float a0 = aw[0], a1 = aw[1], a2 = aw[2], a3 = aw[3];
    const float mx = fmaxf(fmaxf(a0, a1), fmaxf(a2, a3));
    const float e0 = __expf(a0 - mx), e1 = __expf(a1 - mx);
    const float e2 = __expf(a2 - mx), e3 = __expf(a3 - mx);
    const float inv = 1.0f / (e0 + e1 + e2 + e3);
    const float wp[4] = { e0 * inv, e1 * inv, e2 * inv, e3 * inv };

    const float* of    = off + (size_t)t * 64 + hh * 8;
    const float* vbase = value + (size_t)b * L * 256 + hh * 32 + lane;

    float acc = 0.f;
    #pragma unroll
    for (int p = 0; p < 4; ++p) {
        // px = loc_x*W - 0.5 collapses exactly to ix + off_x
        const float px = (float)ix + of[2 * p];
        const float py = (float)iy + of[2 * p + 1];
        const float x0f = floorf(px), y0f = floorf(py);
        const float wx = px - x0f, wy = py - y0f;
        const int x0 = (int)x0f, y0 = (int)y0f;
        const float cw[4] = { (1.f - wx) * (1.f - wy), wx * (1.f - wy),
                              (1.f - wx) * wy,         wx * wy };
        const int xs[4] = { x0, x0 + 1, x0,     x0 + 1 };
        const int ys[4] = { y0, y0,     y0 + 1, y0 + 1 };
        float s = 0.f;
        #pragma unroll
        for (int c = 0; c < 4; ++c) {
            const int xx = xs[c], yy = ys[c];
            const bool valid = (xx >= 0) & (xx < Ww) & (yy >= 0) & (yy < Hh);
            const int cx = min(max(xx, 0), Ww - 1);
            const int cy = min(max(yy, 0), Hh - 1);
            const float v = vbase[(size_t)(cy * Ww + cx) * 256];
            s += valid ? cw[c] * v : 0.f;
        }
        acc += wp[p] * s;
    }
    outh[(size_t)t * 256 + hh * 32 + lane] = (_Float16)acc;
}

// ---------------------------------------------------------------------------
// Fused residual add + LayerNorm (D=256): one wave32 per row, shuffle reduce.
// ---------------------------------------------------------------------------
__global__ __launch_bounds__(256)
void add_ln_kernel(const float* __restrict__ xa, const float* __restrict__ xb,
                   const float* __restrict__ g,  const float* __restrict__ be,
                   float* __restrict__ y, _Float16* __restrict__ yh, int rows)
{
    const int gid  = blockIdx.x * blockDim.x + threadIdx.x;
    const int lane = gid & 31;
    const int row  = gid >> 5;
    if (row >= rows) return;
    const size_t base = (size_t)row * 256;

    float v[8];
    float s = 0.f;
    #pragma unroll
    for (int i = 0; i < 8; ++i) {
        const int d = lane + i * 32;
        v[i] = xa[base + d] + xb[base + d];
        s += v[i];
    }
    #pragma unroll
    for (int o = 16; o > 0; o >>= 1) s += __shfl_xor(s, o, 32);
    const float mu = s * (1.0f / 256.0f);
    float var = 0.f;
    #pragma unroll
    for (int i = 0; i < 8; ++i) { const float d = v[i] - mu; var += d * d; }
    #pragma unroll
    for (int o = 16; o > 0; o >>= 1) var += __shfl_xor(var, o, 32);
    const float rs = rsqrtf(var * (1.0f / 256.0f) + 1e-5f);
    #pragma unroll
    for (int i = 0; i < 8; ++i) {
        const int d = lane + i * 32;
        const float o = (v[i] - mu) * rs * g[d] + be[d];
        y[base + d] = o;
        if (yh) yh[base + d] = (_Float16)o;
    }
}

// ---------------------------------------------------------------------------
__global__ void f2h_kernel(const float* __restrict__ in, _Float16* __restrict__ out, int n)
{
    const int i = blockIdx.x * blockDim.x + threadIdx.x;
    if (i < n) out[i] = (_Float16)in[i];
}

// ---------------------------------------------------------------------------
extern "C" void kernel_launch(void* const* d_in, const int* in_sizes, int n_in,
                              void* d_out, int out_size, void* d_ws, size_t ws_size,
                              hipStream_t stream)
{
    const int Bb = 2, Hh = 128, Ww = 128, D = 256, DFF = 2048;
    const int M = Bb * Hh * Ww;               // 32768 tokens
    (void)in_sizes; (void)n_in; (void)out_size; (void)ws_size;

    const float* src    = (const float*)d_in[0];
    const float* W_off  = (const float*)d_in[1];
    const float* b_off  = (const float*)d_in[2];
    const float* W_attn = (const float*)d_in[3];
    const float* b_attn = (const float*)d_in[4];
    const float* W_val  = (const float*)d_in[5];
    const float* b_val  = (const float*)d_in[6];
    const float* W_out  = (const float*)d_in[7];
    const float* b_out  = (const float*)d_in[8];
    const float* W1     = (const float*)d_in[9];
    const float* b1     = (const float*)d_in[10];
    const float* W2     = (const float*)d_in[11];
    const float* b2     = (const float*)d_in[12];
    const float* g1     = (const float*)d_in[13];
    const float* be1    = (const float*)d_in[14];
    const float* g2     = (const float*)d_in[15];
    const float* be2    = (const float*)d_in[16];

    char* ws = (char*)d_ws;
    size_t cur = 0;
    auto alloc = [&](size_t bytes) -> char* {
        char* p = ws + cur;
        cur += (bytes + 255) & ~(size_t)255;
        return p;
    };
    _Float16* xh     = (_Float16*)alloc((size_t)M * D * 2);      // x f16, reused as h1 f16
    _Float16* wvh    = (_Float16*)alloc((size_t)D * D * 2);
    _Float16* woffh  = (_Float16*)alloc((size_t)64 * D * 2);
    _Float16* wattnh = (_Float16*)alloc((size_t)32 * D * 2);
    _Float16* wouth  = (_Float16*)alloc((size_t)D * D * 2);
    _Float16* w1h    = (_Float16*)alloc((size_t)DFF * D * 2);
    _Float16* w2h    = (_Float16*)alloc((size_t)D * DFF * 2);
    float*    bufA   = (float*)   alloc((size_t)M * D * 4);      // value -> src2 -> ffn2
    float*    offbuf = (float*)   alloc((size_t)M * 64 * 4);
    float*    attnbuf= (float*)   alloc((size_t)M * 32 * 4);
    _Float16* outh   = (_Float16*)alloc((size_t)M * D * 2);
    float*    h1     = (float*)   alloc((size_t)M * D * 4);
    _Float16* f1h    = (_Float16*)alloc((size_t)M * DFF * 2);
    _Float16* h1h    = xh;                                       // reuse after projections

    // f32 -> f16 conversions
    f2h_kernel<<<(M * D + 255) / 256, 256, 0, stream>>>(src, xh, M * D);
    f2h_kernel<<<(D * D + 255) / 256, 256, 0, stream>>>(W_val, wvh, D * D);
    f2h_kernel<<<(64 * D + 255) / 256, 256, 0, stream>>>(W_off, woffh, 64 * D);
    f2h_kernel<<<(32 * D + 255) / 256, 256, 0, stream>>>(W_attn, wattnh, 32 * D);
    f2h_kernel<<<(D * D + 255) / 256, 256, 0, stream>>>(W_out, wouth, D * D);
    f2h_kernel<<<(DFF * D + 255) / 256, 256, 0, stream>>>(W1, w1h, DFF * D);
    f2h_kernel<<<(D * DFF + 255) / 256, 256, 0, stream>>>(W2, w2h, D * DFF);

    const dim3 blk(256);
    const int MB = M / BM;   // 256

    // Projections: value [M,256], off [M,64], attn [M,32]
    gemm_wmma_kernel<false, true, false><<<dim3(D / BN, MB), blk, 0, stream>>>(
        xh, wvh, b_val, bufA, nullptr, M, D, D);
    gemm_wmma_kernel<false, true, false><<<dim3(1, MB), blk, 0, stream>>>(
        xh, woffh, b_off, offbuf, nullptr, M, 64, D);
    gemm_wmma_kernel<false, true, false><<<dim3(1, MB), blk, 0, stream>>>(
        xh, wattnh, b_attn, attnbuf, nullptr, M, 32, D);

    // Deformable sampling: M blocks x 256 threads = one wave per (token, head)
    msda_sample_kernel<<<M, 256, 0, stream>>>(bufA, offbuf, attnbuf, outh, Bb, Hh, Ww);

    // src2 = out @ W_out^T + b_out  (into bufA, value no longer needed)
    gemm_wmma_kernel<false, true, false><<<dim3(D / BN, MB), blk, 0, stream>>>(
        outh, wouth, b_out, bufA, nullptr, M, D, D);

    // h1 = LN(x + src2); also f16 copy for FFN
    add_ln_kernel<<<M / 8, 256, 0, stream>>>(src, bufA, g1, be1, h1, h1h, M);

    // FFN1: relu(h1 @ W1^T + b1) -> f16 [M, 2048]
    gemm_wmma_kernel<true, false, true><<<dim3(DFF / BN, MB), blk, 0, stream>>>(
        h1h, w1h, b1, nullptr, f1h, M, DFF, D);

    // FFN2: f1 @ W2^T + b2 -> f32 (into bufA, src2 no longer needed)
    gemm_wmma_kernel<false, true, false><<<dim3(D / BN, MB), blk, 0, stream>>>(
        f1h, w2h, b2, bufA, nullptr, M, D, DFF);

    // out = LN(h1 + ffn2)
    add_ln_kernel<<<M / 8, 256, 0, stream>>>(h1, bufA, g2, be2, (float*)d_out, nullptr, M);
}
